// Model1_52656299049197
// MI455X (gfx1250) — matrix operand order repaired and checked by reading, not compiled, not run on hardware
//
#include <hip/hip_runtime.h>

// ---------------------------------------------------------------------------
// Types for CDNA5 WMMA (gfx1250, wave32)
// ---------------------------------------------------------------------------
typedef __bf16 bf16;
typedef __attribute__((ext_vector_type(16))) __bf16 v16bf;
typedef __attribute__((ext_vector_type(8)))  float  v8f;

#define HD   2048
#define HD3  6144
#define TOKD 1024
#define TT   512
#define NLAB 50000
#define MAXL 50

// ---------------------------------------------------------------------------
// Helpers
// ---------------------------------------------------------------------------
__device__ __forceinline__ bf16 f2bf(float f) {
    unsigned u = __float_as_uint(f);
    u += 0x7fffu + ((u >> 16) & 1u);            // round to nearest even
    unsigned short s = (unsigned short)(u >> 16);
    return __builtin_bit_cast(bf16, s);
}
__device__ __forceinline__ float sigmoidf_(float x) { return 1.0f / (1.0f + __expf(-x)); }

// Two contiguous 16B loads -> 16 packed bf16 fragment elements.
__device__ __forceinline__ v16bf load_frag(const bf16* p0, const bf16* p1) {
    union { uint4 u; bf16 h[8]; } a, b;
    a.u = *(const uint4*)p0;
    b.u = *(const uint4*)p1;
    v16bf f;
#pragma unroll
    for (int i = 0; i < 8; ++i) { f[i] = a.h[i]; f[8 + i] = b.h[i]; }
    return f;
}
__device__ __forceinline__ v16bf zero_frag() {
    v16bf f;
#pragma unroll
    for (int i = 0; i < 16; ++i) f[i] = (bf16)0.0f;
    return f;
}
// A-matrix fragment for M=1 matvec: vector v lives in row M=0 (lanes 0 and 16),
// all other rows zero. Lane group g takes K [kc+16g, +16) contiguous, matching
// the B fragments.
__device__ __forceinline__ v16bf vec_frag(const bf16* v, int kc, int lane) {
    int lm = lane & 15, g = lane >> 4;
    if (lm == 0) return load_frag(v + kc + 16 * g, v + kc + 16 * g + 8);
    return zero_frag();
}
__device__ __forceinline__ v8f wmma_bf(v16bf a, v16bf b, v8f c) {
    return __builtin_amdgcn_wmma_f32_16x16x32_bf16(false, a, false, b, (short)0, c, false, false);
}

// ---------------------------------------------------------------------------
// Init: zero h buffers / sync vars / argmax slot (ws is poisoned by harness)
// ---------------------------------------------------------------------------
__global__ void k_init(float* hbuf, bf16* hbbf, unsigned long long* amax, int* sync) {
    int i = blockIdx.x * blockDim.x + threadIdx.x;
    if (i < 4 * HD) { hbuf[i] = 0.0f; hbbf[i] = (bf16)0.0f; }
    if (i == 0) { amax[0] = 0ull; sync[0] = 0; sync[1] = 0; }
}

// ---------------------------------------------------------------------------
// f32 -> bf16 weight conversion (grid-stride)
// ---------------------------------------------------------------------------
__global__ void k_cvt_bf16(const float* __restrict__ src, bf16* __restrict__ dst, long long n) {
    long long i = (long long)blockIdx.x * blockDim.x + threadIdx.x;
    long long st = (long long)gridDim.x * blockDim.x;
    for (; i < n; i += st) dst[i] = f2bf(src[i]);
}

// Embedding gather + convert: emb_bf[t][k] = bf16(E[x[t]][k])
__global__ void k_gather_emb(const int* __restrict__ x, const float* __restrict__ E,
                             bf16* __restrict__ emb) {
    int i = blockIdx.x * blockDim.x + threadIdx.x;   // 512*1024 threads
    int t = i >> 10, k = i & 1023;
    emb[i] = f2bf(E[(long long)x[t] * TOKD + k]);
}

// ---------------------------------------------------------------------------
// Batched input-side GEMM: gi[t][j] = sum_k emb_src[t][k]*Wih[j][k] + bih[j]
// M=512, N=6144, K=1024. One wave per 16x16 C tile. reverseA=1 reads emb[511-t].
// ---------------------------------------------------------------------------
__global__ void k_gemm_gi(const bf16* __restrict__ A, const bf16* __restrict__ W,
                          const float* __restrict__ bih, float* __restrict__ gi,
                          int reverseA) {
    int lane = threadIdx.x & 31;
    int wid  = (blockIdx.x * blockDim.x + threadIdx.x) >> 5;  // 0..12287
    int mt = wid & 31, nt = wid >> 5;
    int m0 = mt * 16, n0 = nt * 16;
    int lm = lane & 15, g = lane >> 4;
    int arow = m0 + lm;
    if (reverseA) arow = TT - 1 - arow;
    const bf16* ap = A + (long long)arow * TOKD;
    const bf16* bp = W + (long long)(n0 + lm) * TOKD;   // B column n is Wih row n (contiguous K)
    v8f acc = {};
    for (int kc = 0; kc < TOKD; kc += 32) {
        v16bf af = load_frag(ap + kc + 16 * g, ap + kc + 16 * g + 8);
        v16bf bfr = load_frag(bp + kc + 16 * g, bp + kc + 16 * g + 8);
        acc = wmma_bf(af, bfr, acc);
    }
    int col = n0 + lm;
    float bv = bih[col];
#pragma unroll
    for (int r = 0; r < 8; ++r) {
        int row = m0 + r + 8 * g;                       // C layout: VGPR r -> M = r + 8*(lane>>4)
        gi[(long long)row * HD3 + col] = acc[r] + bv;
    }
}

// ---------------------------------------------------------------------------
// Persistent encoder: both GRU directions, 512 sequential steps with grid sync.
// 32 blocks x 256 thr = 256 waves; wave owns 16 channels j of one direction and
// computes all three gate rows (j, j+2048, j+4096) via 3 WMMA accumulators with
// h broadcast in A row 0 -> lane l<16 holds (gh_r, gh_z, gh_n) for j=j0+l.
// ---------------------------------------------------------------------------
__global__ void k_encoder(const bf16* __restrict__ WhhF, const bf16* __restrict__ WhhB,
                          const float* __restrict__ giF, const float* __restrict__ giB,
                          const float* __restrict__ bhhF, const float* __restrict__ bhhB,
                          float* __restrict__ hbuf, bf16* __restrict__ hbbf, int* sync) {
    const int NB = 32;
    int lane = threadIdx.x & 31;
    int wg   = blockIdx.x * 8 + (threadIdx.x >> 5);   // 0..255
    int dir  = wg >> 7;                               // 0=fwd, 1=bwd
    int j0   = (wg & 127) * 16;
    const bf16*  W   = dir ? WhhB : WhhF;
    const float* gi  = dir ? giB  : giF;
    const float* bhh = dir ? bhhB : bhhF;
    float* hb0  = hbuf + dir * 2 * HD;                // [2][2048] double buffer
    bf16*  hbb0 = hbbf + dir * 2 * HD;
    int lm = lane & 15, g = lane >> 4;
    const bf16* bpR = W + (long long)(j0 + lm) * HD;
    const bf16* bpZ = W + (long long)(j0 + lm + HD) * HD;
    const bf16* bpN = W + (long long)(j0 + lm + 2 * HD) * HD;
    float bR = 0.f, bZ = 0.f, bN = 0.f;
    if (lane < 16) {
        bR = bhh[j0 + lane]; bZ = bhh[j0 + lane + HD]; bN = bhh[j0 + lane + 2 * HD];
    }
    for (int t = 0; t < TT; ++t) {
        int cur = t & 1, nxt = cur ^ 1;
        const bf16* hv = hbb0 + cur * HD;
        v8f aR = {}, aZ = {}, aN = {};
        for (int kc = 0; kc < HD; kc += 32) {
            v16bf af = vec_frag(hv, kc, lane);
            v16bf b1 = load_frag(bpR + kc + 16 * g, bpR + kc + 16 * g + 8);
            v16bf b2 = load_frag(bpZ + kc + 16 * g, bpZ + kc + 16 * g + 8);
            v16bf b3 = load_frag(bpN + kc + 16 * g, bpN + kc + 16 * g + 8);
            aR = wmma_bf(af, b1, aR);
            aZ = wmma_bf(af, b2, aZ);
            aN = wmma_bf(af, b3, aN);
        }
        if (lane < 16) {
            int j = j0 + lane;
            const float* git = gi + (long long)t * HD3;
            float r  = sigmoidf_(git[j]          + aR[0] + bR);
            float z  = sigmoidf_(git[j + HD]     + aZ[0] + bZ);
            float n  = tanhf    (git[j + 2 * HD] + r * (aN[0] + bN));
            float hp = hb0[cur * HD + j];
            float hn = (1.0f - z) * n + z * hp;
            hb0 [nxt * HD + j] = hn;
            hbb0[nxt * HD + j] = f2bf(hn);
        }
        if (t != TT - 1) {                             // grid-wide barrier
            __threadfence();
            __syncthreads();
            if (threadIdx.x == 0) {
                int prev = atomicAdd(sync, 1);
                if (prev == NB * (t + 1) - 1) {
                    atomicExch(sync + 1, t + 1);
                } else {
                    volatile int* genp = sync + 1;
                    while (*genp < t + 1) __builtin_amdgcn_s_sleep(2);
                }
            }
            __syncthreads();
            __threadfence();
        }
    }
}

// ---------------------------------------------------------------------------
// sent = concat(hF,hB) @ W_mid.T + b_mid   (one-time f32 wave-dot)
// ---------------------------------------------------------------------------
__global__ void k_mid(const float* __restrict__ hF, const float* __restrict__ hB,
                      const float* __restrict__ Wmid, const float* __restrict__ bmid,
                      float* __restrict__ sent) {
    int lane = threadIdx.x & 31;
    int w = (blockIdx.x * blockDim.x + threadIdx.x) >> 5;   // 0..511
    for (int row = w; row < HD; row += 512) {
        const float* wr = Wmid + (long long)row * (2 * HD);
        float acc = 0.f;
        for (int k = lane * 4; k < 2 * HD; k += 128) {
            float4 wv = *(const float4*)(wr + k);
            float4 xv = (k < HD) ? *(const float4*)(hF + k) : *(const float4*)(hB + (k - HD));
            acc += wv.x * xv.x + wv.y * xv.y + wv.z * xv.z + wv.w * xv.w;
        }
        for (int m = 16; m; m >>= 1) acc += __shfl_xor(acc, m, 32);
        if (lane == 0) sent[row] = acc + bmid[row];
    }
}

// lscores -> d_out[0..49]; carry = concat(zeros, sent) in f32 + bf16
__global__ void k_len_carry(const float* __restrict__ sent, const float* __restrict__ Wlen,
                            const float* __restrict__ blen, float* __restrict__ out_ls,
                            float* __restrict__ carry, bf16* __restrict__ carrybf) {
    if (blockIdx.x == 0) {
        int lane = threadIdx.x & 31;
        int w = threadIdx.x >> 5;
        for (int row = w; row < MAXL; row += 8) {
            const float* wr = Wlen + (long long)row * HD;
            float acc = 0.f;
            for (int k = lane * 4; k < HD; k += 128) {
                float4 wv = *(const float4*)(wr + k);
                float4 xv = *(const float4*)(sent + k);
                acc += wv.x * xv.x + wv.y * xv.y + wv.z * xv.z + wv.w * xv.w;
            }
            for (int m = 16; m; m >>= 1) acc += __shfl_xor(acc, m, 32);
            if (lane == 0) out_ls[row] = acc + blen[row];
        }
    } else {
        int i = (blockIdx.x - 1) * 256 + threadIdx.x;       // 0..2047
        carry[i] = 0.f; carrybf[i] = (bf16)0.0f;
        float s = sent[i];
        carry[HD + i] = s; carrybf[HD + i] = f2bf(s);
    }
}

// ---------------------------------------------------------------------------
// Decoder gates: gi = carry @ Wih_d.T (+bih_d); h-side preact is just bhh_d.
// WMMA matvec, K=4096. 128 waves (16 blocks x 256).
// ---------------------------------------------------------------------------
__global__ void k_dec_gates(const bf16* __restrict__ Wihd, const bf16* __restrict__ carrybf,
                            const float* __restrict__ bihd, const float* __restrict__ bhhd,
                            float* __restrict__ hdec, bf16* __restrict__ hdecbf,
                            unsigned long long* amax) {
    int lane = threadIdx.x & 31;
    int w = (blockIdx.x * blockDim.x + threadIdx.x) >> 5;   // 0..127
    int j0 = w * 16;
    int lm = lane & 15, g = lane >> 4;
    const bf16* bpR = Wihd + (long long)(j0 + lm) * (2 * HD);
    const bf16* bpZ = Wihd + (long long)(j0 + lm + HD) * (2 * HD);
    const bf16* bpN = Wihd + (long long)(j0 + lm + 2 * HD) * (2 * HD);
    v8f aR = {}, aZ = {}, aN = {};
    for (int kc = 0; kc < 2 * HD; kc += 32) {
        v16bf af = vec_frag(carrybf, kc, lane);
        aR = wmma_bf(af, load_frag(bpR + kc + 16 * g, bpR + kc + 16 * g + 8), aR);
        aZ = wmma_bf(af, load_frag(bpZ + kc + 16 * g, bpZ + kc + 16 * g + 8), aZ);
        aN = wmma_bf(af, load_frag(bpN + kc + 16 * g, bpN + kc + 16 * g + 8), aN);
    }
    if (lane < 16) {
        int j = j0 + lane;
        float r = sigmoidf_(aR[0] + bihd[j]          + bhhd[j]);
        float z = sigmoidf_(aZ[0] + bihd[j + HD]     + bhhd[j + HD]);
        float n = tanhf    (aN[0] + bihd[j + 2 * HD] + r * bhhd[j + 2 * HD]);
        float h = (1.0f - z) * n;
        hdec[j] = h; hdecbf[j] = f2bf(h);
    }
    if (blockIdx.x == 0 && threadIdx.x == 0) amax[0] = 0ull;  // reset argmax for this step
}

// Classifier: scores = h @ W_last.T -> d_out slice; fused argmax via packed u64
// atomicMax (one atomic per wave). Ties resolve to smallest row (jnp.argmax).
__global__ void k_dec_cls(const bf16* __restrict__ Wlast, const bf16* __restrict__ hbf,
                          float* __restrict__ out_scores, unsigned long long* amax) {
    int lane = threadIdx.x & 31;
    int tile = (blockIdx.x * blockDim.x + threadIdx.x) >> 5;
    if (tile >= NLAB / 16) return;                            // wave-uniform guard
    int r0 = tile * 16;
    int lm = lane & 15, g = lane >> 4;
    const bf16* bp = Wlast + (long long)(r0 + lm) * HD;
    v8f acc = {};
    for (int kc = 0; kc < HD; kc += 32) {
        v16bf af = vec_frag(hbf, kc, lane);
        acc = wmma_bf(af, load_frag(bp + kc + 16 * g, bp + kc + 16 * g + 8), acc);
    }
    unsigned long long packed = 0ull;
    if (lane < 16) {
        int row = r0 + lane;
        float s = acc[0];
        out_scores[row] = s;
        unsigned u = __float_as_uint(s);
        u ^= ((int)u >> 31) | 0x80000000u;                    // order-preserving float->uint
        packed = ((unsigned long long)u << 32) | (unsigned)(~row);
    }
    for (int m = 16; m; m >>= 1) {
        unsigned lo = (unsigned)packed, hi = (unsigned)(packed >> 32);
        lo = __shfl_xor(lo, m, 32); hi = __shfl_xor(hi, m, 32);
        unsigned long long other = ((unsigned long long)hi << 32) | lo;
        if (other > packed) packed = other;
    }
    if (lane == 0) atomicMax(amax, packed);
}

// Finalize: decode argmax, emit pred, carry = concat(h, L[pred]) in f32 + bf16.
__global__ void k_dec_fin(const unsigned long long* __restrict__ amax,
                          const float* __restrict__ hdec, const float* __restrict__ L,
                          float* __restrict__ carry, bf16* __restrict__ carrybf,
                          float* __restrict__ out_pred) {
    unsigned long long p = *amax;
    int row = (int)(~(unsigned)p);
    int i = blockIdx.x * blockDim.x + threadIdx.x;            // 0..2047
    float h = hdec[i];
    carry[i] = h;          carrybf[i] = f2bf(h);
    float lv = L[(long long)row * HD + i];
    carry[HD + i] = lv;    carrybf[HD + i] = f2bf(lv);
    if (i == 0) out_pred[0] = (float)row;
}

// ---------------------------------------------------------------------------
// Host orchestration
// ---------------------------------------------------------------------------
extern "C" void kernel_launch(void* const* d_in, const int* in_sizes, int n_in,
                              void* d_out, int out_size, void* d_ws, size_t ws_size,
                              hipStream_t stream) {
    const int*   x      = (const int*)  d_in[0];
    const float* E      = (const float*)d_in[1];
    const float* L      = (const float*)d_in[2];
    const float* Wih_f  = (const float*)d_in[3];
    const float* Whh_f  = (const float*)d_in[4];
    const float* bih_f  = (const float*)d_in[5];
    const float* bhh_f  = (const float*)d_in[6];
    const float* Wih_b  = (const float*)d_in[7];
    const float* Whh_b  = (const float*)d_in[8];
    const float* bih_b  = (const float*)d_in[9];
    const float* bhh_b  = (const float*)d_in[10];
    const float* W_mid  = (const float*)d_in[11];
    const float* b_mid  = (const float*)d_in[12];
    const float* W_len  = (const float*)d_in[13];
    const float* b_len  = (const float*)d_in[14];
    const float* Wih_d  = (const float*)d_in[15];
    /* Whh_d (d_in[16]) drops out of the reference decoder: h0=0 each step */
    const float* bih_d  = (const float*)d_in[17];
    const float* bhh_d  = (const float*)d_in[18];
    const float* W_last = (const float*)d_in[19];
    float* out = (float*)d_out;

    char* ws = (char*)d_ws;
    size_t off = 0;
    auto carve = [&](size_t bytes) -> char* {
        char* p = ws + off;
        off = (off + bytes + 255) & ~((size_t)255);
        return p;
    };
    bf16* emb_bf   = (bf16*)carve((size_t)TT * TOKD * 2);
    bf16* WihF_bf  = (bf16*)carve((size_t)HD3 * TOKD * 2);
    bf16* WihB_bf  = (bf16*)carve((size_t)HD3 * TOKD * 2);
    bf16* WhhF_bf  = (bf16*)carve((size_t)HD3 * HD * 2);
    bf16* WhhB_bf  = (bf16*)carve((size_t)HD3 * HD * 2);
    bf16* Wihd_bf  = (bf16*)carve((size_t)HD3 * 2 * HD * 2);
    bf16* Wlast_bf = (bf16*)carve((size_t)NLAB * HD * 2);
    float* giF     = (float*)carve((size_t)TT * HD3 * 4);
    float* giB     = (float*)carve((size_t)TT * HD3 * 4);
    float* hbuf    = (float*)carve((size_t)4 * HD * 4);   // [dir][parity][2048]
    bf16*  hbbf    = (bf16*) carve((size_t)4 * HD * 2);
    float* sent    = (float*)carve((size_t)HD * 4);
    float* carry   = (float*)carve((size_t)2 * HD * 4);
    bf16*  carrybf = (bf16*) carve((size_t)2 * HD * 2);
    float* hdec    = (float*)carve((size_t)HD * 4);
    bf16*  hdecbf  = (bf16*) carve((size_t)HD * 2);
    unsigned long long* amax = (unsigned long long*)carve(64);
    int* sync      = (int*)  carve(64);
    (void)ws_size; (void)in_sizes; (void)n_in; (void)out_size;

    // 0. init small state
    k_init<<<32, 256, 0, stream>>>(hbuf, hbbf, amax, sync);

    // 1. bf16 weight copies
    k_cvt_bf16<<<2048, 256, 0, stream>>>(Wih_f,  WihF_bf,  (long long)HD3 * TOKD);
    k_cvt_bf16<<<2048, 256, 0, stream>>>(Wih_b,  WihB_bf,  (long long)HD3 * TOKD);
    k_cvt_bf16<<<2048, 256, 0, stream>>>(Whh_f,  WhhF_bf,  (long long)HD3 * HD);
    k_cvt_bf16<<<2048, 256, 0, stream>>>(Whh_b,  WhhB_bf,  (long long)HD3 * HD);
    k_cvt_bf16<<<4096, 256, 0, stream>>>(Wih_d,  Wihd_bf,  (long long)HD3 * 2 * HD);
    k_cvt_bf16<<<8192, 256, 0, stream>>>(W_last, Wlast_bf, (long long)NLAB * HD);

    // 2. embedding gather (+convert)
    k_gather_emb<<<2048, 256, 0, stream>>>(x, E, emb_bf);

    // 3. time-batched input GEMMs (WMMA): gi = emb @ Wih.T + bih
    k_gemm_gi<<<1536, 256, 0, stream>>>(emb_bf, WihF_bf, bih_f, giF, 0);
    k_gemm_gi<<<1536, 256, 0, stream>>>(emb_bf, WihB_bf, bih_b, giB, 1);

    // 4. persistent bidirectional GRU recurrence (512 steps, grid-synced)
    k_encoder<<<32, 256, 0, stream>>>(WhhF_bf, WhhB_bf, giF, giB, bhh_f, bhh_b,
                                      hbuf, hbbf, sync);

    // 5. sentence vector + length scores + initial decoder carry
    k_mid<<<64, 256, 0, stream>>>(hbuf /*hF parity0*/, hbuf + 2 * HD /*hB parity0*/,
                                  W_mid, b_mid, sent);
    k_len_carry<<<9, 256, 0, stream>>>(sent, W_len, b_len, out, carry, carrybf);

    // 6. decoder: 50 sequential steps
    float* scores_base = out + MAXL;
    float* preds_base  = out + MAXL + (size_t)MAXL * NLAB;
    for (int s = 0; s < MAXL; ++s) {
        k_dec_gates<<<16, 256, 0, stream>>>(Wihd_bf, carrybf, bih_d, bhh_d,
                                            hdec, hdecbf, amax);
        k_dec_cls<<<(NLAB / 16 + 7) / 8, 256, 0, stream>>>(Wlast_bf, hdecbf,
                                                           scores_base + (size_t)s * NLAB, amax);
        k_dec_fin<<<8, 256, 0, stream>>>(amax, hdec, L, carry, carrybf, preds_base + s);
    }
}
